// SingleDecoder_15710990369165
// MI455X (gfx1250) — compile-verified, matrix operand order checked by reading, not run on hardware
//
#include <hip/hip_runtime.h>
#include <cstddef>
#include <cstdint>

#define DSPEC 552
#define DPRE  256
#define DH    1024
#define GSZ   4096
#define LSEQ  400
#define BATCH 64

typedef __attribute__((ext_vector_type(16))) __bf16 v16bf;
typedef __attribute__((ext_vector_type(8)))  float  v8f;

union Frag16 { uint4 u[2]; v16bf v; };

static __device__ __forceinline__ unsigned short f32_to_bf16(float f) {
  unsigned int u = __float_as_uint(f);
  unsigned int r = u + 0x7fffu + ((u >> 16) & 1u);   // round to nearest even
  return (unsigned short)(r >> 16);
}

static __device__ __forceinline__ float sigmoidf_(float x) {
  return 1.0f / (1.0f + __expf(-x));
}
static __device__ __forceinline__ float tanhf_(float x) {
  return 1.0f - 2.0f / (__expf(2.0f * x) + 1.0f);
}

// Load a 16x32 bf16 tile (rows x K) from a row-major matrix with leading dim ld.
// CDNA5 16-bit A/B operand layout: lane L holds row (L&15); half-wave selects
// K sub-blocks {0..7,16..23} vs {8..15,24..31}. Two 16B loads per lane.
static __device__ __forceinline__ v16bf load_tile16x32(const unsigned short* M, int ld,
                                                       int row0, int k0, int rowMax) {
  int lane = threadIdx.x & 31;
  int r = row0 + (lane & 15);
  if (r >= rowMax) r = rowMax - 1;      // clamp; result masked at store
  int kb = (lane >> 4) << 3;            // 0 or 8
  const unsigned short* p = M + (size_t)r * ld + k0 + kb;
  Frag16 f;
  f.u[0] = *reinterpret_cast<const uint4*>(p);       // K = kb..kb+7
  f.u[1] = *reinterpret_cast<const uint4*>(p + 16);  // K = 16+kb..23+kb
  return f.v;
}

static __device__ __forceinline__ v8f wmma_bf16f32(v16bf a, v16bf b, v8f c) {
  return __builtin_amdgcn_wmma_f32_16x16x32_bf16(false, a, false, b, (short)0, c, false, false);
}

// ---------------------------------------------------------------- prep kernels

__global__ void sd_cvt_rows(const float* __restrict__ src, unsigned short* __restrict__ dst,
                            int rows, int src_ld, int cols) {
  size_t i = (size_t)blockIdx.x * blockDim.x + threadIdx.x;
  size_t total = (size_t)rows * cols;
  if (i >= total) return;
  int r = (int)(i / cols);
  int c = (int)(i % cols);
  dst[i] = f32_to_bf16(src[(size_t)r * src_ld + c]);
}

__global__ void sd_add_vec(const float* __restrict__ a, const float* __restrict__ b,
                           float* __restrict__ o, int n) {
  int i = blockIdx.x * blockDim.x + threadIdx.x;
  if (i < n) o[i] = a[i] + b[i];
}

__global__ void sd_zero_u16(unsigned short* p, size_t n) {
  size_t i = (size_t)blockIdx.x * blockDim.x + threadIdx.x;
  if (i < n) p[i] = 0;
}
__global__ void sd_zero_f32(float* p, size_t n) {
  size_t i = (size_t)blockIdx.x * blockDim.x + threadIdx.x;
  if (i < n) p[i] = 0.0f;
}

// ------------------------------------------------------------------- pre-net
// One workgroup per (t,b) row: 552->256 ReLU, 256->256 ReLU, emit bf16.
__global__ void sd_prenet(const float* __restrict__ S_pad,
                          const float* __restrict__ W1, const float* __restrict__ b1,
                          const float* __restrict__ W2, const float* __restrict__ b2,
                          unsigned short* __restrict__ pre2b) {
  __shared__ float xrow[DSPEC];
  __shared__ float h1s[DPRE];
  int r = blockIdx.x;                  // 0..L*B-1
  int t = r / BATCH;
  int b = r % BATCH;
  int tid = threadIdx.x;               // 256 threads
  const float* src = (t == 0) ? nullptr : (S_pad + ((size_t)(t - 1) * BATCH + b) * DSPEC);
  for (int k = tid; k < DSPEC; k += DPRE) xrow[k] = src ? src[k] : 0.0f;
  __syncthreads();
  float s = b1[tid];
  const float* w = W1 + (size_t)tid * DSPEC;
  for (int k = 0; k < DSPEC; ++k) s += w[k] * xrow[k];
  h1s[tid] = fmaxf(s, 0.0f);
  __syncthreads();
  float s2 = b2[tid];
  const float* w2 = W2 + (size_t)tid * DPRE;
  for (int k = 0; k < DPRE; ++k) s2 += w2[k] * h1s[k];
  pre2b[(size_t)r * DPRE + tid] = f32_to_bf16(fmaxf(s2, 0.0f));
}

// ----------------------------------------------------------- layer-0 loaders
// Virtual K space of DPRE+DH: first the input contribution, then recurrent.
static __device__ __forceinline__ v16bf l0_load_a(const unsigned short* pre2_t,
                                                  const unsigned short* h0_prev,
                                                  int row0, int k) {
  return (k < DPRE) ? load_tile16x32(pre2_t, DPRE, row0, k, BATCH)
                    : load_tile16x32(h0_prev, DH, row0, k - DPRE, BATCH);
}
static __device__ __forceinline__ v16bf l0_load_b(const unsigned short* Wih0b,
                                                  const unsigned short* Whh0b,
                                                  int wrow, int k) {
  return (k < DPRE) ? load_tile16x32(Wih0b, DPRE, wrow, k, GSZ)
                    : load_tile16x32(Whh0b, DH, wrow, k - DPRE, GSZ);
}

// -------------------------------------------------------------- LSTM layer 0
// gates = pre2[t] @ Wih0[:, :256]^T + h0_prev @ Whh0^T + (bih0+bhh0); cell -> h0,c0
// grid: 128 WGs (16 h-cols x 32-row group each), 256 threads = 8 waves:
// wave = (row-tile in {0,1}) x (gate in {0..3}); one accumulator per wave.
// Distance-2 software pipeline over the merged K=1280 loop.
__global__ void sd_lstm_step_l0(const unsigned short* __restrict__ pre2_t,   // [64 x 256]
                                const unsigned short* __restrict__ h0_prev,  // [64 x 1024]
                                const unsigned short* __restrict__ Wih0b,    // [4096 x 256]
                                const unsigned short* __restrict__ Whh0b,    // [4096 x 1024]
                                const float* __restrict__ bias0,             // [4096]
                                float* __restrict__ c0,                      // [64 x 1024]
                                unsigned short* __restrict__ h0_out) {       // [64 x 1024]
  __shared__ float gbuf[4][32][17];    // [gate][local row][col], padded stride 17
  int tid  = threadIdx.x;
  int wave = tid >> 5;                 // 0..7
  int lane = tid & 31;
  int mtile = wave & 1;                // local 16-row tile
  int gate  = wave >> 1;               // 0..3
  int r0   = (blockIdx.x & 1) * 32;    // global row base for this WG
  int row0 = r0 + mtile * 16;
  int hc0  = (blockIdx.x >> 1) * 16;   // 16 h-columns
  int wrow = gate * DH + hc0;          // weight row base for this gate tile

  v8f acc;
  for (int e = 0; e < 8; ++e) acc[e] = 0.0f;

  const int KT = DPRE + DH;            // 1280 -> 40 iterations of 32
  v16bf ab[3], bb[3];
  ab[0] = l0_load_a(pre2_t, h0_prev, row0, 0);
  bb[0] = l0_load_b(Wih0b, Whh0b, wrow, 0);
  ab[1] = l0_load_a(pre2_t, h0_prev, row0, 32);
  bb[1] = l0_load_b(Wih0b, Whh0b, wrow, 32);
#pragma unroll
  for (int i = 0; i < KT / 32; ++i) {
    int kp = (i + 2) * 32;
    if (kp < KT) {
      ab[(i + 2) % 3] = l0_load_a(pre2_t, h0_prev, row0, kp);
      bb[(i + 2) % 3] = l0_load_b(Wih0b, Whh0b, wrow, kp);
    }
    acc = wmma_bf16f32(ab[i % 3], bb[i % 3], acc);
  }

  int n = lane & 15;
  int mb = (lane >> 4) * 8;
  float bv = bias0[gate * DH + hc0 + n];
  for (int v = 0; v < 8; ++v)
    gbuf[gate][mtile * 16 + mb + v][n] = acc[v] + bv;
  __syncthreads();

  // ---- Cell nonlinearity: 32 rows x 16 cols, 2 elements per thread.
  for (int e = tid; e < 32 * 16; e += 256) {
    int lr = e >> 4;
    int col = e & 15;
    size_t idx = (size_t)(r0 + lr) * DH + hc0 + col;
    float gi = gbuf[0][lr][col];
    float gf = gbuf[1][lr][col];
    float gg = gbuf[2][lr][col];
    float go = gbuf[3][lr][col];
    float c = sigmoidf_(gf) * c0[idx] + sigmoidf_(gi) * tanhf_(gg);
    c0[idx] = c;
    h0_out[idx] = f32_to_bf16(sigmoidf_(go) * tanhf_(c));
  }
}

// -------------------------------------------------------------- LSTM layer 1
// gates = h0 @ Wih1^T + h1_prev @ Whh1^T + (bih1+bhh1); cell -> h1 (stored in Hb[t])
// Distance-2 software pipeline over both K=1024 streams.
__global__ void sd_lstm_step_l1(const unsigned short* __restrict__ h0_cur,   // [64 x 1024]
                                const unsigned short* __restrict__ h1_prev,  // [64 x 1024]
                                const unsigned short* __restrict__ Wih1b,    // [4096 x 1024]
                                const unsigned short* __restrict__ Whh1b,    // [4096 x 1024]
                                const float* __restrict__ bias1,             // [4096]
                                float* __restrict__ c1,                      // [64 x 1024]
                                unsigned short* __restrict__ h1_out) {       // [64 x 1024]
  __shared__ float gbuf[4][32][17];
  int tid  = threadIdx.x;
  int wave = tid >> 5;
  int lane = tid & 31;
  int mtile = wave & 1;
  int gate  = wave >> 1;
  int r0   = (blockIdx.x & 1) * 32;
  int row0 = r0 + mtile * 16;
  int hc0  = (blockIdx.x >> 1) * 16;
  int wrow = gate * DH + hc0;

  v8f acc;
  for (int e = 0; e < 8; ++e) acc[e] = 0.0f;

  v16bf a0b[3], b0b[3], a1b[3], b1b[3];
  a0b[0] = load_tile16x32(h0_cur,  DH, row0, 0, BATCH);
  b0b[0] = load_tile16x32(Wih1b,   DH, wrow, 0, GSZ);
  a1b[0] = load_tile16x32(h1_prev, DH, row0, 0, BATCH);
  b1b[0] = load_tile16x32(Whh1b,   DH, wrow, 0, GSZ);
  a0b[1] = load_tile16x32(h0_cur,  DH, row0, 32, BATCH);
  b0b[1] = load_tile16x32(Wih1b,   DH, wrow, 32, GSZ);
  a1b[1] = load_tile16x32(h1_prev, DH, row0, 32, BATCH);
  b1b[1] = load_tile16x32(Whh1b,   DH, wrow, 32, GSZ);
#pragma unroll
  for (int i = 0; i < DH / 32; ++i) {
    int kp = (i + 2) * 32;
    if (kp < DH) {
      int s = (i + 2) % 3;
      a0b[s] = load_tile16x32(h0_cur,  DH, row0, kp, BATCH);
      b0b[s] = load_tile16x32(Wih1b,   DH, wrow, kp, GSZ);
      a1b[s] = load_tile16x32(h1_prev, DH, row0, kp, BATCH);
      b1b[s] = load_tile16x32(Whh1b,   DH, wrow, kp, GSZ);
    }
    acc = wmma_bf16f32(a0b[i % 3], b0b[i % 3], acc);
    acc = wmma_bf16f32(a1b[i % 3], b1b[i % 3], acc);
  }

  int n = lane & 15;
  int mb = (lane >> 4) * 8;
  float bv = bias1[gate * DH + hc0 + n];
  for (int v = 0; v < 8; ++v)
    gbuf[gate][mtile * 16 + mb + v][n] = acc[v] + bv;
  __syncthreads();

  for (int e = tid; e < 32 * 16; e += 256) {
    int lr = e >> 4;
    int col = e & 15;
    size_t idx = (size_t)(r0 + lr) * DH + hc0 + col;
    float gi = gbuf[0][lr][col];
    float gf = gbuf[1][lr][col];
    float gg = gbuf[2][lr][col];
    float go = gbuf[3][lr][col];
    float c = sigmoidf_(gf) * c1[idx] + sigmoidf_(gi) * tanhf_(gg);
    c1[idx] = c;
    h1_out[idx] = f32_to_bf16(sigmoidf_(go) * tanhf_(c));
  }
}

// ------------------------------------------------ generic WMMA GEMM (+ bias)
// D[M x N] fp32 = A[M x K]_bf16 @ W[N x K]_bf16^T + bias[N]
// grid: (M/64, ceil(N/64)), block 128 (4 waves -> 4 M-tiles; 4 N-tiles each).
__global__ void sd_wmma_gemm_bias(const unsigned short* __restrict__ A,
                                  const unsigned short* __restrict__ W,
                                  const float* __restrict__ bias,
                                  float* __restrict__ D,
                                  int M, int N, int K) {
  int wave = threadIdx.x >> 5;
  int lane = threadIdx.x & 31;
  int row0 = blockIdx.x * 64 + wave * 16;
  int col0 = blockIdx.y * 64;
  v8f acc[4];
  for (int j = 0; j < 4; ++j)
    for (int e = 0; e < 8; ++e) acc[j][e] = 0.0f;

  for (int k0 = 0; k0 < K; k0 += 32) {
    v16bf a = load_tile16x32(A, K, row0, k0, M);
    for (int j = 0; j < 4; ++j) {
      v16bf w = load_tile16x32(W, K, col0 + j * 16, k0, N);
      acc[j] = wmma_bf16f32(a, w, acc[j]);
    }
  }

  int n = lane & 15;
  int mb = (lane >> 4) * 8;
  for (int j = 0; j < 4; ++j) {
    int col = col0 + j * 16 + n;
    if (col >= N) continue;
    float bv = bias ? bias[col] : 0.0f;
    for (int v = 0; v < 8; ++v) {
      int row = row0 + mb + v;
      if (row < M) D[(size_t)row * N + col] = acc[j][v] + bv;
    }
  }
}

// --------------------------------------------------------------------- host

extern "C" void kernel_launch(void* const* d_in, const int* in_sizes, int n_in,
                              void* d_out, int out_size, void* d_ws, size_t ws_size,
                              hipStream_t stream) {
  (void)in_sizes; (void)n_in; (void)out_size; (void)ws_size;
  const float* S_pad = (const float*)d_in[0];
  const float* W1    = (const float*)d_in[1];
  const float* b1    = (const float*)d_in[2];
  const float* W2    = (const float*)d_in[3];
  const float* b2    = (const float*)d_in[4];
  const float* Wih0  = (const float*)d_in[5];
  const float* Whh0  = (const float*)d_in[6];
  const float* bih0  = (const float*)d_in[7];
  const float* bhh0  = (const float*)d_in[8];
  const float* Wih1  = (const float*)d_in[9];
  const float* Whh1  = (const float*)d_in[10];
  const float* bih1  = (const float*)d_in[11];
  const float* bhh1  = (const float*)d_in[12];
  const float* Wo    = (const float*)d_in[13];
  const float* bo    = (const float*)d_in[14];
  float* out = (float*)d_out;

  char* ws = (char*)d_ws;
  size_t off = 0;
  auto take = [&](size_t bytes) -> char* {
    char* p = ws + off;
    off += (bytes + 255) & ~(size_t)255;
    return p;
  };
  unsigned short* Wih0b  = (unsigned short*)take((size_t)GSZ * DPRE * 2);
  unsigned short* Whh0b  = (unsigned short*)take((size_t)GSZ * DH * 2);
  unsigned short* Wih1b  = (unsigned short*)take((size_t)GSZ * DH * 2);
  unsigned short* Whh1b  = (unsigned short*)take((size_t)GSZ * DH * 2);
  unsigned short* Wob    = (unsigned short*)take((size_t)DSPEC * DH * 2);
  float*          bias0  = (float*)take((size_t)GSZ * 4);
  float*          bias1  = (float*)take((size_t)GSZ * 4);
  unsigned short* pre2b  = (unsigned short*)take((size_t)LSEQ * BATCH * DPRE * 2);
  unsigned short* h0buf  = (unsigned short*)take((size_t)2 * BATCH * DH * 2);
  unsigned short* h1zero = (unsigned short*)take((size_t)BATCH * DH * 2);
  float*          c0     = (float*)take((size_t)BATCH * DH * 4);
  float*          c1     = (float*)take((size_t)BATCH * DH * 4);
  unsigned short* Hb     = (unsigned short*)take((size_t)LSEQ * BATCH * DH * 2);

  // Weight conversion to bf16 (drop zero-context columns of Wih0 / Wo).
  {
    size_t n;
    n = (size_t)GSZ * DPRE;
    sd_cvt_rows<<<dim3((unsigned)((n + 255) / 256)), 256, 0, stream>>>(Wih0, Wih0b, GSZ, DPRE + 128, DPRE);
    n = (size_t)GSZ * DH;
    sd_cvt_rows<<<dim3((unsigned)((n + 255) / 256)), 256, 0, stream>>>(Whh0, Whh0b, GSZ, DH, DH);
    sd_cvt_rows<<<dim3((unsigned)((n + 255) / 256)), 256, 0, stream>>>(Wih1, Wih1b, GSZ, DH, DH);
    sd_cvt_rows<<<dim3((unsigned)((n + 255) / 256)), 256, 0, stream>>>(Whh1, Whh1b, GSZ, DH, DH);
    n = (size_t)DSPEC * DH;
    sd_cvt_rows<<<dim3((unsigned)((n + 255) / 256)), 256, 0, stream>>>(Wo, Wob, DSPEC, DH + 128, DH);
  }
  sd_add_vec<<<dim3(GSZ / 256), 256, 0, stream>>>(bih0, bhh0, bias0, GSZ);
  sd_add_vec<<<dim3(GSZ / 256), 256, 0, stream>>>(bih1, bhh1, bias1, GSZ);

  // Deterministic zero-init of recurrent state (re-runs on every graph replay).
  sd_zero_u16<<<dim3((2 * BATCH * DH + 255) / 256), 256, 0, stream>>>(h0buf, (size_t)2 * BATCH * DH);
  sd_zero_u16<<<dim3((BATCH * DH + 255) / 256), 256, 0, stream>>>(h1zero, (size_t)BATCH * DH);
  sd_zero_f32<<<dim3((BATCH * DH + 255) / 256), 256, 0, stream>>>(c0, (size_t)BATCH * DH);
  sd_zero_f32<<<dim3((BATCH * DH + 255) / 256), 256, 0, stream>>>(c1, (size_t)BATCH * DH);

  // Pre-net over all L*B rows (parallel).
  sd_prenet<<<dim3(LSEQ * BATCH), 256, 0, stream>>>(S_pad, W1, b1, W2, b2, pre2b);

  // Sequential recurrence: 2 kernels per step; weights stay hot in L2.
  for (int t = 0; t < LSEQ; ++t) {
    const unsigned short* pre2_t = pre2b + (size_t)t * BATCH * DPRE;
    unsigned short* h0_out = h0buf + (size_t)(t & 1) * BATCH * DH;
    const unsigned short* h0_in = h0buf + (size_t)((t & 1) ^ 1) * BATCH * DH;
    const unsigned short* h1_in = (t == 0) ? h1zero : (Hb + (size_t)(t - 1) * BATCH * DH);
    unsigned short* h1_out = Hb + (size_t)t * BATCH * DH;
    sd_lstm_step_l0<<<dim3(128), 256, 0, stream>>>(pre2_t, h0_in, Wih0b, Whh0b, bias0, c0, h0_out);
    sd_lstm_step_l1<<<dim3(128), 256, 0, stream>>>(h0_out, h1_in, Wih1b, Whh1b, bias1, c1, h1_out);
  }

  // Output projection: [25600 x 1024] @ [552 x 1024]^T + bo.
  sd_wmma_gemm_bias<<<dim3(LSEQ * BATCH / 64, (DSPEC + 63) / 64), 128, 0, stream>>>(
      Hb, Wob, bo, out, LSEQ * BATCH, DSPEC, DH);
}
